// SinkhornWarpInterpolator_87720412054032
// MI455X (gfx1250) — matrix operand order repaired and checked by reading, not compiled, not run on hardware
//
#include <hip/hip_runtime.h>
#include <hip/hip_bf16.h>

typedef __attribute__((ext_vector_type(16))) _Float16 v16h;
typedef __attribute__((ext_vector_type(8)))  float    v8f;

#define NB 16          // batch
#define HP 64          // token map H
#define WP 64          // token map W
#define DD 128         // token dim
#define NK 405         // SE(2) candidates
#define WIN 8
#define NWIN 64        // WIN*WIN
#define SINK_ITERS 20
#define DUSTBIN -2.0f
#define LOGIT_SCALE 1.7677669529663689f  // 1/(sqrt(128)*0.05)

// ---------------------------------------------------------------------------
// SE(2) candidate parameters: k = a*81 + iy*9 + ix, angle = (a*5-10) deg
// ---------------------------------------------------------------------------
__device__ __forceinline__ void se2_params(int k, float& cth, float& sth,
                                           float& tx, float& ty) {
  const float d2r = 0.017453292519943295f;
  float th = (float)((k / 81) * 5 - 10) * d2r;
  __sincosf(th, &sth, &cth);
  float dyv = (float)(((k / 9) % 9) - 4);
  float dxv = (float)((k % 9) - 4);
  tx = dxv * (2.0f / 63.0f);
  ty = dyv * (2.0f / 63.0f);
}

// ---------------------------------------------------------------------------
// K1: patch tokens + L2 normalize + channel-mean score map.
// One wave (32 lanes) per token; 8 tokens per 256-thread block.
// outH != nullptr -> write f16 tokens (f0 path); else write f32 (f1 path).
// ---------------------------------------------------------------------------
__global__ void __launch_bounds__(256)
patch_token_kernel(const float* __restrict__ z, _Float16* __restrict__ outH,
                   float* __restrict__ outF, float* __restrict__ outS) {
  int wave = threadIdx.x >> 5, lane = threadIdx.x & 31;
  int tok = blockIdx.x * 8 + wave;          // 0 .. NB*4096-1
  int b  = tok >> 12;
  int p  = tok & 4095;
  int hp = p >> 6, wp = p & 63;

  float v[4];
  float ss = 0.f;
#pragma unroll
  for (int kk = 0; kk < 4; ++kk) {
    int d = kk * 32 + lane;
    int c = d >> 4, py = (d >> 2) & 3, px = d & 3;
    float x = z[(((size_t)b * 8 + c) * 256 + (hp * 4 + py)) * 256 + (wp * 4 + px)];
    v[kk] = x;
    ss += x * x;
  }
#pragma unroll
  for (int o = 16; o; o >>= 1) ss += __shfl_xor(ss, o, 32);
  float inv = 1.0f / fmaxf(sqrtf(ss), 1e-6f);

  float sm = 0.f;
#pragma unroll
  for (int kk = 0; kk < 4; ++kk) {
    float nv = v[kk] * inv;
    sm += nv;
    size_t oidx = (size_t)tok * DD + kk * 32 + lane;
    if (outH) outH[oidx] = (_Float16)nv;
    else      outF[oidx] = nv;
  }
#pragma unroll
  for (int o = 16; o; o >>= 1) sm += __shfl_xor(sm, o, 32);
  if (lane == 0) outS[tok] = sm * (1.0f / 128.0f);
}

// ---------------------------------------------------------------------------
// K2: score one SE(2) candidate per block: mean over pixels of
//     grid_sample_zeros(f1s, grid_k) * f0s.
// ---------------------------------------------------------------------------
__global__ void __launch_bounds__(256)
se2_score_kernel(const float* __restrict__ f0s, const float* __restrict__ f1s,
                 float* __restrict__ score) {
  __shared__ float red[256];
  int b = blockIdx.x / NK;
  int k = blockIdx.x % NK;
  float cth, sth, tx, ty;
  se2_params(k, cth, sth, tx, ty);

  const float* F0 = f0s + (size_t)b * 4096;
  const float* F1 = f1s + (size_t)b * 4096;
  float acc = 0.f;
  for (int p = threadIdx.x; p < 4096; p += 256) {
    int py = p >> 6, px = p & 63;
    float gx = -1.0f + px * (2.0f / 63.0f);
    float gy = -1.0f + py * (2.0f / 63.0f);
    float gX = cth * gx - sth * gy + tx;
    float gY = sth * gx + cth * gy + ty;
    float ix = (gX + 1.0f) * 0.5f * 63.0f;
    float iy = (gY + 1.0f) * 0.5f * 63.0f;
    float x0 = floorf(ix), y0 = floorf(iy);
    float wx1 = ix - x0, wx0 = 1.0f - wx1;
    float wy1 = iy - y0, wy0 = 1.0f - wy1;
    float val = 0.f;
#pragma unroll
    for (int cy = 0; cy < 2; ++cy)
#pragma unroll
      for (int cx = 0; cx < 2; ++cx) {
        float yy = y0 + cy, xx = x0 + cx;
        if (yy >= 0.f && yy <= 63.f && xx >= 0.f && xx <= 63.f) {
          float w = (cy ? wy1 : wy0) * (cx ? wx1 : wx0);
          val += F1[(int)yy * 64 + (int)xx] * w;
        }
      }
    acc += val * F0[p];
  }
  red[threadIdx.x] = acc;
  __syncthreads();
  for (int s = 128; s > 0; s >>= 1) {
    if (threadIdx.x < s) red[threadIdx.x] += red[threadIdx.x + s];
    __syncthreads();
  }
  if (threadIdx.x == 0) score[b * NK + k] = red[0] * (1.0f / 4096.0f);
}

// ---------------------------------------------------------------------------
// K3: per-sample argmax over 405 candidates (first-max tie break).
// ---------------------------------------------------------------------------
__global__ void __launch_bounds__(256)
se2_argmax_kernel(const float* __restrict__ score, int* __restrict__ bestk) {
  __shared__ float bv[256];
  __shared__ int   bi[256];
  int b = blockIdx.x;
  float best = -1e30f;
  int   idx  = 0;
  for (int k = threadIdx.x; k < NK; k += 256) {
    float s = score[b * NK + k];
    if (s > best) { best = s; idx = k; }
  }
  bv[threadIdx.x] = best;
  bi[threadIdx.x] = idx;
  __syncthreads();
  for (int s = 128; s > 0; s >>= 1) {
    if (threadIdx.x < s) {
      float ov = bv[threadIdx.x + s];
      int   oi = bi[threadIdx.x + s];
      if (ov > bv[threadIdx.x] || (ov == bv[threadIdx.x] && oi < bi[threadIdx.x])) {
        bv[threadIdx.x] = ov;
        bi[threadIdx.x] = oi;
      }
    }
    __syncthreads();
  }
  if (threadIdx.x == 0) bestk[b] = bi[0];
}

// ---------------------------------------------------------------------------
// K4: apply best SE(2) warp to the 128-channel f1 token map; emit f16.
// One block per output pixel; thread d handles channel d (coalesced).
// ---------------------------------------------------------------------------
__global__ void __launch_bounds__(128)
apply_se2_kernel(const float* __restrict__ f1f, const int* __restrict__ bestk,
                 _Float16* __restrict__ f1ah) {
  int blk = blockIdx.x;
  int b = blk >> 12, p = blk & 4095;
  int py = p >> 6, px = p & 63;
  float cth, sth, tx, ty;
  se2_params(bestk[b], cth, sth, tx, ty);

  float gx = -1.0f + px * (2.0f / 63.0f);
  float gy = -1.0f + py * (2.0f / 63.0f);
  float gX = cth * gx - sth * gy + tx;
  float gY = sth * gx + cth * gy + ty;
  float ix = (gX + 1.0f) * 0.5f * 63.0f;
  float iy = (gY + 1.0f) * 0.5f * 63.0f;
  float x0 = floorf(ix), y0 = floorf(iy);
  float wx1 = ix - x0, wx0 = 1.0f - wx1;
  float wy1 = iy - y0, wy0 = 1.0f - wy1;

  int d = threadIdx.x;
  float acc = 0.f;
#pragma unroll
  for (int cy = 0; cy < 2; ++cy)
#pragma unroll
    for (int cx = 0; cx < 2; ++cx) {
      float yy = y0 + cy, xx = x0 + cx;
      if (yy >= 0.f && yy <= 63.f && xx >= 0.f && xx <= 63.f) {
        float w = (cy ? wy1 : wy0) * (cx ? wx1 : wx0);
        acc += w * f1f[((size_t)b * 4096 + (int)yy * 64 + (int)xx) * DD + d];
      }
    }
  f1ah[((size_t)b * 4096 + p) * DD + d] = (_Float16)acc;
}

// ---------------------------------------------------------------------------
// K5: fused per-window logits GEMM (v_wmma_f32_16x16x32_f16) + log-domain
// Sinkhorn (20 iters, dustbin) + expected-coordinate delta / confidence.
// One 256-thread block (8 wave32) per 8x8 window; 1024 blocks total.
// ---------------------------------------------------------------------------
__global__ void __launch_bounds__(256)
sinkhorn_wmma_kernel(const _Float16* __restrict__ f0h,
                     const _Float16* __restrict__ f1ah,
                     float* __restrict__ out) {
  __shared__ __align__(16) _Float16 Xs[NWIN * DD];   // 16 KB
  __shared__ __align__(16) _Float16 Ys[NWIN * DD];   // 16 KB
  __shared__ float LP[65 * 66];                      // padded stride 66 (bank-safe)

  int blk = blockIdx.x;            // 0..1023
  int b  = blk >> 6;
  int i0 = (blk >> 3) & 7;
  int j0 = blk & 7;
  int tid = threadIdx.x;
  int lane = tid & 31, wave = tid >> 5;

  // ---- stage window tokens into LDS (uint4 = 8 halves) --------------------
  for (int q = tid; q < NWIN * DD / 8; q += 256) {   // 1024 uint4 per matrix
    int n = q >> 4, dch = q & 15;
    int R = i0 * 8 + (n >> 3), C = j0 * 8 + (n & 7);
    size_t g = (size_t)((b << 12) + R * 64 + C) * (DD / 8) + dch;
    ((uint4*)Xs)[q] = ((const uint4*)f0h)[g];
    ((uint4*)Ys)[q] = ((const uint4*)f1ah)[g];
  }
  __syncthreads();

  // ---- 64x64 logits via 16 WMMA tiles: wave w -> tiles (ti, tj0), (ti, tj0+1)
  int h = lane >> 4;               // lane half (ISA fragment layouts)
  int N = lane & 15;
  int ti  = wave >> 1;             // 0..3
  int tj0 = (wave & 1) * 2;        // 0 or 2
  int Arow = ti * 16 + (lane & 15);
  int Bm0  = tj0 * 16 + N;
  int Bm1  = Bm0 + 16;

  v8f c0 = {}, c1 = {};
#pragma unroll
  for (int kt = 0; kt < 4; ++kt) { // K = 128 in chunks of 32
    v16h a, b0, b1;
    // A 16x32 f16: lane -> 32 contiguous bytes: K in [16h, 16h+16)
    uint4* ap = (uint4*)&a;
    ap[0] = *(const uint4*)&Xs[Arow * DD + kt * 32 + 16 * h];
    ap[1] = *(const uint4*)&Xs[Arow * DD + kt * 32 + 16 * h + 8];
    // B 32x16 f16: lane -> 32 contiguous bytes: K in [16h, 16h+16)
    uint4* b0p = (uint4*)&b0;
    b0p[0] = *(const uint4*)&Ys[Bm0 * DD + kt * 32 + 16 * h];
    b0p[1] = *(const uint4*)&Ys[Bm0 * DD + kt * 32 + 16 * h + 8];
    uint4* b1p = (uint4*)&b1;
    b1p[0] = *(const uint4*)&Ys[Bm1 * DD + kt * 32 + 16 * h];
    b1p[1] = *(const uint4*)&Ys[Bm1 * DD + kt * 32 + 16 * h + 8];
    c0 = __builtin_amdgcn_wmma_f32_16x16x32_f16(false, a, false, b0, (short)0, c0, false, false);
    c1 = __builtin_amdgcn_wmma_f32_16x16x32_f16(false, a, false, b1, (short)0, c1, false, false);
  }

  // C layout: lane, VGPR v -> (M = v + 8h, N = lane&15)
  int Mbase = ti * 16 + 8 * h;
#pragma unroll
  for (int v = 0; v < 8; ++v) {
    LP[(Mbase + v) * 66 + tj0 * 16 + N]       = c0[v] * LOGIT_SCALE;
    LP[(Mbase + v) * 66 + (tj0 + 1) * 16 + N] = c1[v] * LOGIT_SCALE;
  }
  __syncthreads();
  if (tid < 65) {                   // dustbin row/col
    LP[tid * 66 + 64] = DUSTBIN;
    LP[64 * 66 + tid] = DUSTBIN;
  }
  __syncthreads();

  // ---- 20 Sinkhorn iterations (row lse then col lse), wave32 reductions ---
  for (int it = 0; it < SINK_ITERS; ++it) {
    for (int r = wave; r < 65; r += 8) {
      float v0 = LP[r * 66 + lane];
      float v1 = LP[r * 66 + 32 + lane];
      float v2 = (lane == 0) ? LP[r * 66 + 64] : -1e30f;
      float m = fmaxf(fmaxf(v0, v1), v2);
#pragma unroll
      for (int o = 16; o; o >>= 1) m = fmaxf(m, __shfl_xor(m, o, 32));
      float s = __expf(v0 - m) + __expf(v1 - m) + ((lane == 0) ? __expf(v2 - m) : 0.f);
#pragma unroll
      for (int o = 16; o; o >>= 1) s += __shfl_xor(s, o, 32);
      float lse = m + __logf(s);
      LP[r * 66 + lane]      = v0 - lse;
      LP[r * 66 + 32 + lane] = v1 - lse;
      if (lane == 0) LP[r * 66 + 64] = v2 - lse;
    }
    __syncthreads();
    for (int c = wave; c < 65; c += 8) {
      float v0 = LP[lane * 66 + c];
      float v1 = LP[(lane + 32) * 66 + c];
      float v2 = (lane == 0) ? LP[64 * 66 + c] : -1e30f;
      float m = fmaxf(fmaxf(v0, v1), v2);
#pragma unroll
      for (int o = 16; o; o >>= 1) m = fmaxf(m, __shfl_xor(m, o, 32));
      float s = __expf(v0 - m) + __expf(v1 - m) + ((lane == 0) ? __expf(v2 - m) : 0.f);
#pragma unroll
      for (int o = 16; o; o >>= 1) s += __shfl_xor(s, o, 32);
      float lse = m + __logf(s);
      LP[lane * 66 + c]        = v0 - lse;
      LP[(lane + 32) * 66 + c] = v1 - lse;
      if (lane == 0) LP[64 * 66 + c] = v2 - lse;
    }
    __syncthreads();
  }

  // ---- expected coords, delta, confidence ---------------------------------
  for (int n = wave; n < NWIN; n += 8) {
    float p0 = __expf(LP[n * 66 + lane]);
    float p1 = __expf(LP[n * 66 + 32 + lane]);
    int m0 = lane, m1 = lane + 32;
    float mass = p0 + p1;
    float sx = p0 * (float)(m0 & 7) + p1 * (float)(m1 & 7);
    float sy = p0 * (float)(m0 >> 3) + p1 * (float)(m1 >> 3);
#pragma unroll
    for (int o = 16; o; o >>= 1) {
      mass += __shfl_xor(mass, o, 32);
      sx   += __shfl_xor(sx, o, 32);
      sy   += __shfl_xor(sy, o, 32);
    }
    if (lane == 0) {
      float mm = fmaxf(mass, 1e-8f);
      int R = i0 * 8 + (n >> 3), C = j0 * 8 + (n & 7);
      size_t o_ = ((size_t)(b << 12) + R * 64 + C) * 3;
      out[o_ + 0] = sx / mm - (float)(n & 7);
      out[o_ + 1] = sy / mm - (float)(n >> 3);
      out[o_ + 2] = mass;
    }
  }
}

// ---------------------------------------------------------------------------
extern "C" void kernel_launch(void* const* d_in, const int* in_sizes, int n_in,
                              void* d_out, int out_size, void* d_ws, size_t ws_size,
                              hipStream_t stream) {
  const float* z0 = (const float*)d_in[0];
  const float* z1 = (const float*)d_in[1];

  char* ws = (char*)d_ws;
  _Float16* f0h  = (_Float16*)(ws);                                  // 16 MiB
  _Float16* f1ah = (_Float16*)(ws + (size_t)16 * 1024 * 1024);       // 16 MiB
  float*    f1f  = (float*)(ws + (size_t)32 * 1024 * 1024);          // 32 MiB
  float*    f0s  = (float*)(ws + (size_t)64 * 1024 * 1024);          // 256 KiB
  float*    f1s  = (float*)(ws + (size_t)64 * 1024 * 1024 + 262144); // 256 KiB
  float*    score= (float*)(ws + (size_t)64 * 1024 * 1024 + 524288); // 26 KiB
  int*      bestk= (int*)  (ws + (size_t)64 * 1024 * 1024 + 524288 + 32768);

  // K1: tokens (f0 -> f16, f1 -> f32) + score maps
  patch_token_kernel<<<NB * 4096 / 8, 256, 0, stream>>>(z0, f0h, nullptr, f0s);
  patch_token_kernel<<<NB * 4096 / 8, 256, 0, stream>>>(z1, nullptr, f1f, f1s);
  // K2/K3: exhaustive SE(2) search + argmax
  se2_score_kernel<<<NB * NK, 256, 0, stream>>>(f0s, f1s, score);
  se2_argmax_kernel<<<NB, 256, 0, stream>>>(score, bestk);
  // K4: warp f1 tokens with best SE(2), emit f16
  apply_se2_kernel<<<NB * 4096, 128, 0, stream>>>(f1f, bestk, f1ah);
  // K5: fused WMMA logits + Sinkhorn + delta/conf
  sinkhorn_wmma_kernel<<<NB * 64, 256, 0, stream>>>(f0h, f1ah, (float*)d_out);
}
